// BioGPTRelationExtractor_39762807226986
// MI455X (gfx1250) — compile-verified, matrix operand order checked by reading, not run on hardware
//
#include <hip/hip_runtime.h>
#include <hip/hip_bf16.h>

#define N_NODES 8192
#define E_EDGES 98304
#define E_TOT   (E_EDGES + N_NODES)   // 106496 with self loops
#define HIDDIM  1024
#define CDIM    256
#define H1      4
#define NEG_SLOPE 0.2f

typedef float  v8f   __attribute__((ext_vector_type(8)));
typedef __bf16 v16bf __attribute__((ext_vector_type(16)));
typedef __bf16 v8bf  __attribute__((ext_vector_type(8)));
typedef unsigned int u32x4 __attribute__((ext_vector_type(4)));
typedef int          i32x4 __attribute__((ext_vector_type(4)));
typedef int          i32x8 __attribute__((ext_vector_type(8)));

// ---------- helpers ----------
__device__ __forceinline__ unsigned short f2bf(float f) {
    unsigned u = __float_as_uint(f);
    unsigned r = u + 0x7FFFu + ((u >> 16) & 1u);   // round-to-nearest-even
    return (unsigned short)(r >> 16);
}
__device__ __forceinline__ int fkey(float f) {
    int i = __float_as_int(f);
    return (i >= 0) ? i : (i ^ 0x7FFFFFFF);
}
__device__ __forceinline__ float fdec(int k) {
    int i = (k >= 0) ? k : (k ^ 0x7FFFFFFF);
    return __int_as_float(i);
}
__device__ __forceinline__ void edge_pair(const int* __restrict__ ei, int e, int& s, int& d) {
    if (e < E_EDGES) { s = ei[e]; d = ei[E_EDGES + e]; }
    else             { s = d = e - E_EDGES; }          // self loop
}
__device__ __forceinline__ v16bf load_frag(const unsigned short* row32, int half) {
    // 16-bit A/B fragment: lane holds K = half*8..half*8+7 and 16+half*8..16+half*8+7
    v8bf lo = *reinterpret_cast<const v8bf*>(row32 + half * 8);
    v8bf hi = *reinterpret_cast<const v8bf*>(row32 + 16 + half * 8);
    return __builtin_shufflevector(lo, hi, 0,1,2,3,4,5,6,7,8,9,10,11,12,13,14,15);
}

// ---------- TDM: tensor_load_to_lds of a 2-D K-contiguous bf16 tile ----------
// D# packing per CDNA5 ISA ch.8: group0 = {count, lds_addr, global_addr, type=2},
// group1 = {data_size=1 (2B), tensor_dim0/1, tile_dim0/1, tensor_dim0_stride}.
__device__ __forceinline__ void tdm_load_tile_bf16(
    unsigned lds_off, const unsigned short* gptr,
    unsigned tensor_k, unsigned tensor_rows,
    unsigned tile_k, unsigned tile_rows)
{
#if __has_builtin(__builtin_amdgcn_tensor_load_to_lds)
    unsigned long long ga = (unsigned long long)(size_t)gptr;
    u32x4 g0;
    g0.x = 1u;                                           // count=1, user mode, no gather
    g0.y = lds_off;                                      // lds_addr [63:32]
    g0.z = (unsigned)(ga & 0xFFFFFFFFull);               // global_addr [95:64]
    g0.w = (unsigned)((ga >> 32) & 0x1FFFFFFull) | (2u << 30); // addr[56:32] | type=2
    i32x8 g1;
    g1[0] = (int)(1u << 16);                             // data_size=1 (2 bytes)
    g1[1] = (int)((tensor_k & 0xFFFFu) << 16);           // tensor_dim0 lo -> bits[63:48]
    g1[2] = (int)((tensor_k >> 16) & 0xFFFFu)            // tensor_dim0 hi
          | (int)((tensor_rows & 0xFFFFu) << 16);        // tensor_dim1 lo
    g1[3] = (int)((tensor_rows >> 16) & 0xFFFFu)         // tensor_dim1 hi
          | (int)((tile_k & 0xFFFFu) << 16);             // tile_dim0
    g1[4] = (int)(tile_rows & 0xFFFFu);                  // tile_dim1 (tile_dim2 = 0)
    g1[5] = (int)tensor_k;                               // tensor_dim0_stride [31:0]
    g1[6] = 0;                                           // stride0 hi / stride1 lo
    g1[7] = 0;
    i32x4 z4 = {0, 0, 0, 0};
#if __clang_major__ >= 23
    i32x8 z8 = {0, 0, 0, 0, 0, 0, 0, 0};
    __builtin_amdgcn_tensor_load_to_lds(g0, g1, z4, z4, z8, 0);
#else
    __builtin_amdgcn_tensor_load_to_lds(g0, g1, z4, z4, 0);
#endif
#else
    (void)lds_off; (void)gptr; (void)tensor_k; (void)tensor_rows;
    (void)tile_k; (void)tile_rows;
#endif
}

__device__ __forceinline__ void tdm_wait() {
#if __has_builtin(__builtin_amdgcn_s_wait_tensorcnt)
    __builtin_amdgcn_s_wait_tensorcnt(0);
#else
    asm volatile("s_wait_tensorcnt 0x0" ::: "memory");
#endif
}

// ---------- fills ----------
__global__ void fill_u32(unsigned* p, unsigned v, int n) {
    int t = blockIdx.x * blockDim.x + threadIdx.x;
    if (t < n) p[t] = v;
}

// ---------- fp32 -> bf16 (element-wise, vectorized x4) ----------
__global__ void conv_bf16(const float* __restrict__ in, unsigned short* __restrict__ out, int n) {
    int t4 = (blockIdx.x * blockDim.x + threadIdx.x) * 4;
    if (t4 >= n) return;
    float4 f = *reinterpret_cast<const float4*>(in + t4);
    uint2 pk;
    pk.x = (unsigned)f2bf(f.x) | ((unsigned)f2bf(f.y) << 16);
    pk.y = (unsigned)f2bf(f.z) | ((unsigned)f2bf(f.w) << 16);
    *reinterpret_cast<uint2*>(out + t4) = pk;
}

// ---------- fp32 W[K,Nn] -> bf16 WT[Nn,K] (transpose for K-contiguous TDM tiles) ----------
__global__ void conv_bf16_T(const float* __restrict__ W, unsigned short* __restrict__ WT,
                            int K, int Nn) {
    int t = blockIdx.x * blockDim.x + threadIdx.x;
    if (t >= K * Nn) return;
    int k = t / Nn, n = t - k * Nn;
    WT[(size_t)n * K + k] = f2bf(W[t]);
}

// ---------- one K-step of WMMA on a staged tile pair ----------
__device__ __forceinline__ void wmma_step(
    const unsigned short* __restrict__ cA, const unsigned short* __restrict__ cB,
    int wm, int wn, int r, int half, v8f (&acc)[2][2])
{
    #pragma unroll
    for (int ti = 0; ti < 2; ++ti) {
        v16bf a = load_frag(&cA[(wm * 32 + ti * 16 + r) * 32], half);
        #pragma unroll
        for (int tj = 0; tj < 2; ++tj) {
            v16bf b = load_frag(&cB[(wn * 32 + tj * 16 + r) * 32], half);
            acc[ti][tj] = __builtin_amdgcn_wmma_f32_16x16x32_bf16(
                false, a, false, b, (short)0, acc[ti][tj], false, false);
        }
    }
}

// ---------- WMMA bf16 GEMM, double-buffered TDM pipeline ----------
// C[M,Nn] = A[M,K] * BT[Nn,K]^T, A/BT bf16 row-major (K contiguous), C fp32.
// block = 256 threads (8 waves). Block tile 128x64; wave tile 32x32 (2x2 wmma).
// Pipeline: wait(tile i) -> barrier -> issue DMA(tile i+1, other buffer) -> WMMA(tile i).
__global__ __launch_bounds__(256) void gemm_bf16_wmma_tdm(
    const unsigned short* __restrict__ A, const unsigned short* __restrict__ BT,
    float* __restrict__ C, int M, int Nn, int K)
{
    __shared__ unsigned short sA[2][128 * 32];  // [buf][m][k]
    __shared__ unsigned short sB[2][64 * 32];   // [buf][n][k]

    const int tid  = threadIdx.x;
    const int lane = tid & 31;
    const int w    = tid >> 5;
    const int wm   = w >> 1;         // 0..3
    const int wn   = w & 1;          // 0..1
    const int bm   = blockIdx.x * 128;
    const int bn   = blockIdx.y * 64;
    const int half = lane >> 4;
    const int r    = lane & 15;
    const int nt   = K >> 5;         // 32-wide K tiles

    v8f acc[2][2] = {};

#if __has_builtin(__builtin_amdgcn_tensor_load_to_lds)
    const unsigned ldsA[2] = { (unsigned)(size_t)(void*)sA[0], (unsigned)(size_t)(void*)sA[1] };
    const unsigned ldsB[2] = { (unsigned)(size_t)(void*)sB[0], (unsigned)(size_t)(void*)sB[1] };

    if (tid < 32) {   // wave 0 drives the tile DMAs
        tdm_load_tile_bf16(ldsA[0], A + (size_t)bm * K, K, M, 32, 128);
        tdm_load_tile_bf16(ldsB[0], BT + (size_t)bn * K, K, Nn, 32, 64);
    }
    for (int i = 0; i < nt; ++i) {
        if (tid < 32) tdm_wait();          // tile i resident
        __syncthreads();                   // (implies each wave's prior ds_loads retired)
        if (tid < 32 && (i + 1) < nt) {    // prefetch tile i+1 under the WMMAs of tile i
            int nb = (i + 1) & 1;
            tdm_load_tile_bf16(ldsA[nb], A + (size_t)bm * K + (i + 1) * 32, K, M, 32, 128);
            tdm_load_tile_bf16(ldsB[nb], BT + (size_t)bn * K + (i + 1) * 32, K, Nn, 32, 64);
        }
        wmma_step(sA[i & 1], sB[i & 1], wm, wn, r, half, acc);
    }
#else
    // fallback: synchronous cooperative staging, two barriers per step
    for (int i = 0; i < nt; ++i) {
        int k0 = i * 32, cb = i & 1;
        #pragma unroll
        for (int j = 0; j < 2; ++j) {
            int c8 = (j * 256 + tid) * 8;
            int row = c8 >> 5, col = c8 & 31;
            *reinterpret_cast<uint4*>(&sA[cb][row * 32 + col]) =
                *reinterpret_cast<const uint4*>(A + (size_t)(bm + row) * K + k0 + col);
        }
        {
            int c8 = tid * 8;
            int row = c8 >> 5, col = c8 & 31;
            *reinterpret_cast<uint4*>(&sB[cb][row * 32 + col]) =
                *reinterpret_cast<const uint4*>(BT + (size_t)(bn + row) * K + k0 + col);
        }
        __syncthreads();
        wmma_step(sA[cb], sB[cb], wm, wn, r, half, acc);
        __syncthreads();
    }
#endif

    // C/D layout: lane -> n = lane&15, vgpr vr -> m = vr + 8*(lane>>4)
    #pragma unroll
    for (int ti = 0; ti < 2; ++ti) {
        int row0 = bm + wm * 32 + ti * 16 + 8 * half;
        #pragma unroll
        for (int tj = 0; tj < 2; ++tj) {
            int col0 = bn + wn * 32 + tj * 16 + r;
            #pragma unroll
            for (int vr = 0; vr < 8; ++vr)
                C[(size_t)(row0 + vr) * Nn + col0] = acc[ti][tj][vr];
        }
    }
}

// ---------- edge scoring: one wave per (edge, head) ----------
template<int H>
__global__ __launch_bounds__(256) void edge_score_k(
    const float* __restrict__ xl, const float* __restrict__ xr,
    const float* __restrict__ att, const int* __restrict__ ei,
    float* __restrict__ esc, int* __restrict__ mkey)
{
    int gw = (blockIdx.x * blockDim.x + threadIdx.x) >> 5;
    int lane = threadIdx.x & 31;
    if (gw >= E_TOT * H) return;
    int e = gw / H, h = gw - e * H;
    int s, d; edge_pair(ei, e, s, d);
    const float* pl = xl + (size_t)s * (H * 256) + h * 256;
    const float* pr = xr + (size_t)d * (H * 256) + h * 256;
    const float* pa = att + h * 256;
    float sum = 0.f;
    #pragma unroll
    for (int i = 0; i < 8; ++i) {
        int c = lane + i * 32;
        float v = pl[c] + pr[c];
        v = (v > 0.f) ? v : NEG_SLOPE * v;     // leaky_relu
        sum = fmaf(v, pa[c], sum);
    }
    #pragma unroll
    for (int off = 16; off > 0; off >>= 1) sum += __shfl_xor(sum, off, 32);
    if (lane == 0) {
        esc[gw] = sum;
        atomicMax(&mkey[d * H + h], fkey(sum));
    }
}

// ---------- exp + denominator ----------
template<int H>
__global__ void edge_exp_k(const int* __restrict__ ei, const int* __restrict__ mkey,
                           float* __restrict__ esc, float* __restrict__ den)
{
    int t = blockIdx.x * blockDim.x + threadIdx.x;
    if (t >= E_TOT * H) return;
    int e = t / H, h = t - e * H;
    int s, d; edge_pair(ei, e, s, d);
    float ex = __expf(esc[t] - fdec(mkey[d * H + h]));
    esc[t] = ex;
    atomicAdd(&den[d * H + h], ex);
}

// ---------- weighted scatter-add: one wave per (edge, head) ----------
template<int H>
__global__ __launch_bounds__(256) void edge_agg_k(
    const int* __restrict__ ei, const float* __restrict__ esc,
    const float* __restrict__ den, const float* __restrict__ xl,
    float* __restrict__ out)
{
    int gw = (blockIdx.x * blockDim.x + threadIdx.x) >> 5;
    int lane = threadIdx.x & 31;
    if (gw >= E_TOT * H) return;
    int e = gw / H, h = gw - e * H;
    int s, d; edge_pair(ei, e, s, d);
    float alpha = esc[gw] / den[d * H + h];
    const float* pl = xl + (size_t)s * (H * 256) + h * 256;
    float* po       = out + (size_t)d * (H * 256) + h * 256;
    #pragma unroll
    for (int i = 0; i < 8; ++i) {
        int c = lane + i * 32;
        atomicAdd(&po[c], alpha * pl[c]);
    }
}

// ---------- bias + relu in place ----------
__global__ void bias_relu_k(float* __restrict__ p, const float* __restrict__ bias,
                            int n, int mask)
{
    int t = blockIdx.x * blockDim.x + threadIdx.x;
    if (t < n) p[t] = fmaxf(p[t] + bias[t & mask], 0.f);
}

// ---------- host ----------
extern "C" void kernel_launch(void* const* d_in, const int* in_sizes, int n_in,
                              void* d_out, int out_size, void* d_ws, size_t ws_size,
                              hipStream_t stream)
{
    const float* x    = (const float*)d_in[0];
    const int*   ei   = (const int*)  d_in[1];
    const float* Wl1  = (const float*)d_in[2];
    const float* Wr1  = (const float*)d_in[3];
    const float* att1 = (const float*)d_in[4];
    const float* b1   = (const float*)d_in[5];
    const float* Wl2  = (const float*)d_in[6];
    const float* Wr2  = (const float*)d_in[7];
    const float* att2 = (const float*)d_in[8];
    const float* b2   = (const float*)d_in[9];
    float* out = (float*)d_out;
    (void)ws_size; (void)in_sizes; (void)n_in; (void)out_size;

    // ---- workspace layout with lifetime-based aliasing (~120 MiB total) ----
    char* base = (char*)d_ws;
    unsigned short* xb   = (unsigned short*)(base + 0);           // 16 MiB, L1-GEMM only
    float* xl1  = (float*)(base + 16777216);                      // 32 MiB
    float* xr1  = (float*)(base + 50331648);                      // 32 MiB
    float* x2   = (float*)(base + 83886080);                      // 32 MiB
    unsigned short* W1lT = (unsigned short*)(base + 117440512);   // 2 MiB  [1024,1024]
    unsigned short* W1rT = (unsigned short*)(base + 119537664);   // 2 MiB
    unsigned short* W2lT = (unsigned short*)(base + 121634816);   // 0.5 MiB [256,1024]
    unsigned short* W2rT = (unsigned short*)(base + 122159104);   // 0.5 MiB
    int*   m1   = (int*)  (base + 122683392);
    float* den1 = (float*)(base + 122814464);
    float* s1   = (float*)(base + 122945536);
    int*   m2   = (int*)  (base + 124649472);
    float* den2 = (float*)(base + 124682240);
    float* s2   = (float*)(base + 124715008);
    // aliases (dead regions reused):
    float* xl2 = (float*)(base + 0);                              // over xb (dead after L1 GEMMs)
    float* xr2 = (float*)(base + 8388608);
    unsigned short* x2b = (unsigned short*)(base + 16777216);     // over xl1 (dead after L1 agg)

    const dim3 blk(256);

    // ===== bf16 staging: activations + transposed weights =====
    conv_bf16<<<(N_NODES * HIDDIM / 4 + 255) / 256, blk, 0, stream>>>(x, xb, N_NODES * HIDDIM);
    conv_bf16_T<<<(HIDDIM * HIDDIM + 255) / 256, blk, 0, stream>>>(Wl1, W1lT, HIDDIM, HIDDIM);
    conv_bf16_T<<<(HIDDIM * HIDDIM + 255) / 256, blk, 0, stream>>>(Wr1, W1rT, HIDDIM, HIDDIM);
    conv_bf16_T<<<(HIDDIM * CDIM + 255) / 256, blk, 0, stream>>>(Wl2, W2lT, HIDDIM, CDIM);
    conv_bf16_T<<<(HIDDIM * CDIM + 255) / 256, blk, 0, stream>>>(Wr2, W2rT, HIDDIM, CDIM);

    // ===== layer 1 GEMMs: xl1 = x@Wl1, xr1 = x@Wr1  (8192x1024x1024) =====
    dim3 g1(N_NODES / 128, HIDDIM / 64);
    gemm_bf16_wmma_tdm<<<g1, blk, 0, stream>>>(xb, W1lT, xl1, N_NODES, HIDDIM, HIDDIM);
    gemm_bf16_wmma_tdm<<<g1, blk, 0, stream>>>(xb, W1rT, xr1, N_NODES, HIDDIM, HIDDIM);

    // ===== layer 1 softmax stats init =====
    fill_u32<<<(N_NODES * H1 + 255) / 256, blk, 0, stream>>>((unsigned*)m1, 0x80000000u, N_NODES * H1);
    fill_u32<<<(N_NODES * H1 + 255) / 256, blk, 0, stream>>>((unsigned*)den1, 0u, N_NODES * H1);
    fill_u32<<<((N_NODES * HIDDIM) + 255) / 256, blk, 0, stream>>>((unsigned*)x2, 0u, N_NODES * HIDDIM);

    // ===== layer 1 edge phase =====
    edge_score_k<H1><<<(E_TOT * H1) / 8, blk, 0, stream>>>(xl1, xr1, att1, ei, s1, m1);
    edge_exp_k<H1><<<(E_TOT * H1 + 255) / 256, blk, 0, stream>>>(ei, m1, s1, den1);
    edge_agg_k<H1><<<(E_TOT * H1) / 8, blk, 0, stream>>>(ei, s1, den1, xl1, x2);
    bias_relu_k<<<((N_NODES * HIDDIM) + 255) / 256, blk, 0, stream>>>(x2, b1, N_NODES * HIDDIM, HIDDIM - 1);

    // ===== stage x2 -> bf16 (xl1 region is dead now) =====
    conv_bf16<<<(N_NODES * HIDDIM / 4 + 255) / 256, blk, 0, stream>>>(x2, x2b, N_NODES * HIDDIM);

    // ===== layer 2 GEMMs: xl2 = x2@Wl2, xr2 = x2@Wr2  (8192x1024x256) =====
    dim3 g2(N_NODES / 128, CDIM / 64);
    gemm_bf16_wmma_tdm<<<g2, blk, 0, stream>>>(x2b, W2lT, xl2, N_NODES, CDIM, HIDDIM);
    gemm_bf16_wmma_tdm<<<g2, blk, 0, stream>>>(x2b, W2rT, xr2, N_NODES, CDIM, HIDDIM);

    // ===== layer 2 softmax stats init + output zero =====
    fill_u32<<<(N_NODES + 255) / 256, blk, 0, stream>>>((unsigned*)m2, 0x80000000u, N_NODES);
    fill_u32<<<(N_NODES + 255) / 256, blk, 0, stream>>>((unsigned*)den2, 0u, N_NODES);
    fill_u32<<<((N_NODES * CDIM) + 255) / 256, blk, 0, stream>>>((unsigned*)out, 0u, N_NODES * CDIM);

    // ===== layer 2 edge phase (heads = 1) =====
    edge_score_k<1><<<E_TOT / 8, blk, 0, stream>>>(xl2, xr2, att2, ei, s2, m2);
    edge_exp_k<1><<<(E_TOT + 255) / 256, blk, 0, stream>>>(ei, m2, s2, den2);
    edge_agg_k<1><<<E_TOT / 8, blk, 0, stream>>>(ei, s2, den2, xl2, out);
    bias_relu_k<<<((N_NODES * CDIM) + 255) / 256, blk, 0, stream>>>(out, b2, N_NODES * CDIM, CDIM - 1);
}
